// CrystalGATVAE_79637283602850
// MI455X (gfx1250) — matrix operand order repaired and checked by reading, not compiled, not run on hardware
//
#include <hip/hip_runtime.h>
#include <math.h>

#define NN 10000
#define NE 160000
#define NG 100
#define HEADS 4
#define NFEAT 4
#define EFEAT 3
#define LATD 64
#define HIDD 128
#define FDIM 1024   // HEADS * 2 * HID

typedef __attribute__((ext_vector_type(16))) __bf16 v16bf;
typedef __attribute__((ext_vector_type(8)))  __bf16 v8bf;
typedef __attribute__((ext_vector_type(8)))  float  v8f;
typedef __attribute__((ext_vector_type(4)))  int    v4i;

#ifndef __has_builtin
#define __has_builtin(x) 0
#endif
#if __has_builtin(__builtin_amdgcn_cvt_pk_bf16_f32)
#define HAVE_PKBF 1
#endif
#if __has_builtin(__builtin_amdgcn_global_load_async_to_lds_b128) && \
    __has_builtin(__builtin_amdgcn_s_wait_asynccnt)
#define HAVE_ASYNC_LDS 1
typedef __attribute__((address_space(1))) v4i* as1_v4i;
typedef __attribute__((address_space(3))) v4i* as3_v4i;
#endif

__device__ __forceinline__ __bf16 f2bf(float x) {
  union { float f; unsigned u; } v; v.f = x;
  unsigned r = (v.u + 0x7FFFu + ((v.u >> 16) & 1u)) >> 16;
  union { unsigned short s; __bf16 b; } o; o.s = (unsigned short)r;
  return o.b;
}

__device__ __forceinline__ void pkstore(__bf16* d, float x, float y) {
#ifdef HAVE_PKBF
  auto r = __builtin_amdgcn_cvt_pk_bf16_f32(x, y);   // v_cvt_pk_bf16_f32
  d[0] = r[0]; d[1] = r[1];
#else
  d[0] = f2bf(x); d[1] = f2bf(y);
#endif
}

__device__ __forceinline__ float siluf(float y) { return y / (1.f + __expf(-y)); }

__device__ __forceinline__ void atomicMaxF(float* addr, float val) {
  unsigned* ua = (unsigned*)addr;
  unsigned old = *ua;
  while (__uint_as_float(old) < val) {
    unsigned assumed = old;
    old = atomicCAS(ua, assumed, __float_as_uint(val));
    if (old == assumed) break;
  }
}

// fp32 -> bf16 streaming conversion (vectorized; one-time per GEMM operand)
__global__ void cvt_bf16_kernel(const float* __restrict__ in, __bf16* __restrict__ out,
                                size_t n) {
  const size_t stride8 = (size_t)gridDim.x * blockDim.x * 8;
  for (size_t i = ((size_t)blockIdx.x * blockDim.x + threadIdx.x) * 8; i < n; i += stride8) {
    if (i + 8 <= n) {
      const float4 f0 = *(const float4*)(in + i);
      const float4 f1 = *(const float4*)(in + i + 4);
      union { v8bf v; __bf16 e[8]; } o;
      pkstore(o.e + 0, f0.x, f0.y); pkstore(o.e + 2, f0.z, f0.w);
      pkstore(o.e + 4, f1.x, f1.y); pkstore(o.e + 6, f1.z, f1.w);
      *(v8bf*)(out + i) = o.v;
    } else {
      for (size_t j = i; j < n; ++j) out[j] = f2bf(in[j]);
    }
  }
}

// ---------------------------------------------------------------------------
// LDS-staged bf16 WMMA GEMM: C[M,N] = act(A[M,K] @ W[K,N] + bias)
// A and W are PRE-CONVERTED bf16 (half the re-read traffic, no cvt in loop).
//
// Block = 256 threads = 8 wave32 computing a 64x64 tile; waves 2(M)x4(N),
// each wave owns 32x16 via two accumulators (2 WMMA reuse one B fragment).
// A tile staged with GLOBAL_LOAD_ASYNC_TO_LDS_B128 when the toolchain has it
// (ASYNCcnt path, no VGPR round-trip), else b128 load + ds_store_b128.
// Bounds by CLAMPING (garbage only reaches guarded stores); K%32 handled by
// one uniform tail step (B zero-padded). Math: 6x ds_load_b128 + 2x v_wmma,
// EXEC all-ones at every WMMA.
//
// Optional gathered-concat A (edge decoder):
//   row m = [ A[idxA[m], 0:Ksplit] , A[idxB[m], 0:K-Ksplit] ]  (Ksplit%8==0)
// ---------------------------------------------------------------------------
__global__ __launch_bounds__(256) void gemm_wmma_kernel(
    const __bf16* __restrict__ A, const __bf16* __restrict__ W,
    const float* __restrict__ bias, float* __restrict__ C,
    int M, int N, int K,
    const int* __restrict__ idxA, const int* __restrict__ idxB,
    int Ksplit, int lda, int act)
{
  __shared__ __align__(16) __bf16 lA[64 * 32];   // [row][k]
  __shared__ __align__(16) __bf16 lB[64 * 32];   // [n][k]   (K-transposed)

  const int tid = threadIdx.x;
  const int m0g = blockIdx.x * 64;
  const int n0g = blockIdx.y * 64;
  const int As  = (lda > 0) ? lda : K;            // A row stride (elements)
  const bool aVec = ((As & 7) == 0);
  const bool bVec = ((N & 7) == 0) && (n0g + 64 <= N);

  // ---- copy-phase roles ----
  const int acRow = tid >> 2;                     // 0..63
  const int acKg  = (tid & 3) * 8;                // 0,8,16,24
  const __bf16* ab0;
  const __bf16* ab1;
  {
    int r = m0g + acRow; if (r >= M) r = M - 1;   // clamp, never guard
    if (idxA) { ab0 = A + (size_t)idxA[r] * As; ab1 = A + (size_t)idxB[r] * As; }
    else      { ab0 = A + (size_t)r * As;       ab1 = ab0; }
  }
  const int bcK  = tid >> 3;                      // 0..31
  const int bcNg = (tid & 7) * 8;                 // 0,8,...,56

  // ---- math-phase roles ----
  const int wave = tid >> 5;
  const int lane = tid & 31;
  const int wm = wave >> 2;                       // 0..1
  const int wn = wave & 3;                        // 0..3
  const int half = lane >> 4;
  const int rl   = lane & 15;
  const int ar0 = (wm * 32 + rl) * 32;
  const int ar1 = ar0 + 16 * 32;
  const int aco = half * 8;
  const int bco = (wn * 16 + rl) * 32 + half * 8;

  v8f acc0 = {}, acc1 = {};

  for (int k0 = 0; k0 < K; k0 += 32) {
    const bool full = (k0 + 32 <= K);             // uniform
    __syncthreads();                              // math done before LDS reuse

    if (full) {
      // ---- A stage ----
      {
        const int kb = k0 + acKg;                 // 8-run cannot straddle Ksplit
        const __bf16* p = (idxA && kb >= Ksplit) ? (ab1 + kb - Ksplit) : (ab0 + kb);
        __bf16* dst = &lA[acRow * 32 + acKg];
        if (aVec) {
#ifdef HAVE_ASYNC_LDS
          __builtin_amdgcn_global_load_async_to_lds_b128(
              (as1_v4i)(unsigned long long)p, (as3_v4i)(unsigned long long)dst,
              /*offset=*/0, /*cpol=*/0);
#else
          *(v8bf*)dst = *(const v8bf*)p;
#endif
        } else {
          v8bf av;
#pragma unroll
          for (int j = 0; j < 8; ++j) av[j] = p[j];
          *(v8bf*)dst = av;
        }
      }
      // ---- B stage: one b128 load + transpose b16 LDS stores ----
      {
        const __bf16* wrow = W + (size_t)(k0 + bcK) * N;
        __bf16* q = &lB[bcNg * 32 + bcK];
        if (bVec) {
          const v8bf bv = *(const v8bf*)(wrow + n0g + bcNg);
          q[0 * 32] = bv[0]; q[1 * 32] = bv[1]; q[2 * 32] = bv[2]; q[3 * 32] = bv[3];
          q[4 * 32] = bv[4]; q[5 * 32] = bv[5]; q[6 * 32] = bv[6]; q[7 * 32] = bv[7];
        } else {
#pragma unroll
          for (int j = 0; j < 8; ++j) {
            int n = n0g + bcNg + j; if (n >= N) n = N - 1;   // clamp
            q[j * 32] = wrow[n];
          }
        }
      }
    } else {
      // ---- K tail (once per GEMM): A clamped, B zero-padded ----
      {
        v8bf av;
#pragma unroll
        for (int j = 0; j < 8; ++j) {
          int k = k0 + acKg + j; if (k >= K) k = K - 1;
          const __bf16* p = (idxA && k >= Ksplit) ? (ab1 + k - Ksplit) : (ab0 + k);
          av[j] = *p;
        }
        *(v8bf*)&lA[acRow * 32 + acKg] = av;
      }
      {
        const int k = k0 + bcK;
        __bf16* q = &lB[bcNg * 32 + bcK];
        if (k < K) {
          const __bf16* wrow = W + (size_t)k * N;
#pragma unroll
          for (int j = 0; j < 8; ++j) {
            int n = n0g + bcNg + j; if (n >= N) n = N - 1;
            q[j * 32] = wrow[n];
          }
        } else {
#pragma unroll
          for (int j = 0; j < 8; ++j) q[j * 32] = f2bf(0.f);
        }
      }
    }
#ifdef HAVE_ASYNC_LDS
    __builtin_amdgcn_s_wait_asynccnt(0);          // my async LDS writes landed
#endif
    __syncthreads();

    // ---- math phase: pure ds_load_b128 + wmma ----
    const v8bf a0lo = *(const v8bf*)&lA[ar0 + aco];
    const v8bf a0hi = *(const v8bf*)&lA[ar0 + 16 + aco];
    const v8bf a1lo = *(const v8bf*)&lA[ar1 + aco];
    const v8bf a1hi = *(const v8bf*)&lA[ar1 + 16 + aco];
    const v8bf blo  = *(const v8bf*)&lB[bco];
    const v8bf bhi  = *(const v8bf*)&lB[bco + 16];
    v16bf a0, a1, b;
#pragma unroll
    for (int i = 0; i < 8; ++i) {
      a0[i] = a0lo[i]; a0[8 + i] = a0hi[i];
      a1[i] = a1lo[i]; a1[8 + i] = a1hi[i];
      b[i]  = blo[i];  b[8 + i]  = bhi[i];
    }
    acc0 = __builtin_amdgcn_wmma_f32_16x16x32_bf16(false, a0, false, b,
                                                   (short)0, acc0, false, false);
    acc1 = __builtin_amdgcn_wmma_f32_16x16x32_bf16(false, a1, false, b,
                                                   (short)0, acc1, false, false);
  }

  // ---- epilogue (only place with real bounds checks) ----
  const int c = n0g + wn * 16 + rl;
  if (c < N) {
    const float bv = bias ? bias[c] : 0.f;
#pragma unroll
    for (int i = 0; i < 8; ++i) {
      const int r0 = m0g + wm * 32 + i + 8 * half;
      if (r0 < M) {
        float y = acc0[i] + bv;
        if (act == 1) y = siluf(y);
        C[(size_t)r0 * N + c] = y;
      }
      const int r1 = r0 + 16;
      if (r1 < M) {
        float y = acc1[i] + bv;
        if (act == 1) y = siluf(y);
        C[(size_t)r1 * N + c] = y;
      }
    }
  }
}

// ---------------------------------------------------------------------------
// Elementwise / scatter kernels (bandwidth-bound segment ops)
// ---------------------------------------------------------------------------
__global__ void fill_kernel(float* p, float v, size_t n) {
  for (size_t i = (size_t)blockIdx.x * blockDim.x + threadIdx.x; i < n;
       i += (size_t)gridDim.x * blockDim.x) p[i] = v;
}

// edge MLP: 3 -> 6 (silu) -> 3, one thread per edge
__global__ void edge_mlp_kernel(const float* __restrict__ ef,
                                const float* __restrict__ W1, const float* __restrict__ b1,
                                const float* __restrict__ W2, const float* __restrict__ b2,
                                float* __restrict__ out, int E) {
  int e = blockIdx.x * blockDim.x + threadIdx.x;
  if (e >= E) return;
  float x[EFEAT], h[2 * EFEAT];
#pragma unroll
  for (int i = 0; i < EFEAT; ++i) x[i] = ef[(size_t)e * EFEAT + i];
#pragma unroll
  for (int j = 0; j < 2 * EFEAT; ++j) {
    float s = b1[j];
#pragma unroll
    for (int i = 0; i < EFEAT; ++i) s += x[i] * W1[i * 2 * EFEAT + j];
    h[j] = siluf(s);
  }
#pragma unroll
  for (int j = 0; j < EFEAT; ++j) {
    float s = b2[j];
#pragma unroll
    for (int i = 0; i < 2 * EFEAT; ++i) s += h[i] * W2[i * EFEAT + j];
    out[(size_t)e * EFEAT + j] = s;
  }
}

// temp[n,:] = seg_sum(concat(proj[src],ef), src) + seg_sum(concat(proj[dst],ef), dst)
// temp rows use padded stride TS (for aligned GEMM A loads)
__global__ void scatter_temp_kernel(const float* __restrict__ proj,
                                    const float* __restrict__ ef2,
                                    const int* __restrict__ src, const int* __restrict__ dst,
                                    float* __restrict__ temp, int E, int din, int TS) {
  const int Wt = din + EFEAT;
  const size_t total = (size_t)E * Wt;
  for (size_t idx = (size_t)blockIdx.x * blockDim.x + threadIdx.x; idx < total;
       idx += (size_t)gridDim.x * blockDim.x) {
    const int e = (int)(idx / Wt);
    const int j = (int)(idx % Wt);
    const int s = src[e], d = dst[e];
    const float efv = (j >= din) ? ef2[(size_t)e * EFEAT + (j - din)] : 0.f;
    const float vs = (j < din) ? proj[(size_t)s * din + j] : efv;
    const float vd = (j < din) ? proj[(size_t)d * din + j] : efv;
    atomicAdd(&temp[(size_t)s * TS + j], vs);
    atomicAdd(&temp[(size_t)d * TS + j], vd);
  }
}

// el/er: per (node, head) dot with attn vectors
__global__ void attn_lr_kernel(const float* __restrict__ hh,
                               const float* __restrict__ al, const float* __restrict__ ar,
                               float* __restrict__ el, float* __restrict__ er,
                               int N, int dout) {
  int t = blockIdx.x * blockDim.x + threadIdx.x;
  if (t >= N * HEADS) return;
  const int n = t / HEADS, h = t % HEADS;
  const int HD = HEADS * dout;
  float sl = 0.f, sr = 0.f;
  for (int d = 0; d < dout; ++d) {
    const float v = hh[(size_t)n * HD + h * dout + d];
    sl += v * al[h * dout + d];
    sr += v * ar[h * dout + d];
  }
  el[t] = sl; er[t] = sr;
}

__global__ void edge_score_kernel(const float* __restrict__ el, const float* __restrict__ er,
                                  const int* __restrict__ src, const int* __restrict__ dst,
                                  float* __restrict__ eatt, float* __restrict__ segmax, int E) {
  int t = blockIdx.x * blockDim.x + threadIdx.x;
  if (t >= E * HEADS) return;
  const int e = t / HEADS, h = t % HEADS;
  float s = el[src[e] * HEADS + h] + er[dst[e] * HEADS + h];
  s = (s > 0.f) ? s : 0.2f * s;               // leaky_relu 0.2
  eatt[t] = s;
  atomicMaxF(&segmax[dst[e] * HEADS + h], s);
}

__global__ void edge_exp_kernel(const float* __restrict__ eatt, const float* __restrict__ segmax,
                                const int* __restrict__ dst,
                                float* __restrict__ eexp, float* __restrict__ segsum, int E) {
  int t = blockIdx.x * blockDim.x + threadIdx.x;
  if (t >= E * HEADS) return;
  const int e = t / HEADS, h = t % HEADS;
  const float ex = __expf(eatt[t] - segmax[dst[e] * HEADS + h]);
  eexp[t] = ex;
  atomicAdd(&segsum[dst[e] * HEADS + h], ex);
}

// out[dst, h*dout+d] += hh[src, h*dout+d] * alpha[e,h]
__global__ void attn_scatter_kernel(const float* __restrict__ hh,
                                    const float* __restrict__ eexp, const float* __restrict__ segsum,
                                    const int* __restrict__ src, const int* __restrict__ dst,
                                    float* __restrict__ out, int E, int dout) {
  const int HD = HEADS * dout;
  const size_t total = (size_t)E * HD;
  for (size_t idx = (size_t)blockIdx.x * blockDim.x + threadIdx.x; idx < total;
       idx += (size_t)gridDim.x * blockDim.x) {
    const int e = (int)(idx / HD);
    const int c = (int)(idx % HD);
    const int h = c / dout;
    const float alpha = eexp[e * HEADS + h] / segsum[dst[e] * HEADS + h];
    atomicAdd(&out[(size_t)dst[e] * HD + c], hh[(size_t)src[e] * HD + c] * alpha);
  }
}

// h_out = silu(layernorm(acc + gat_bias + resid) * g + b) ; one wave32 per row
__global__ __launch_bounds__(128) void ln_silu_kernel(
    const float* __restrict__ acc, const float* __restrict__ resid,
    const float* __restrict__ gb, const float* __restrict__ g, const float* __restrict__ b,
    float* __restrict__ out, int N, int D) {
  const int row = blockIdx.x * 4 + (threadIdx.x >> 5);
  const int lane = threadIdx.x & 31;
  if (row >= N) return;
  const float* a = acc + (size_t)row * D;
  const float* r = resid + (size_t)row * D;
  float s = 0.f, s2 = 0.f;
  for (int j = lane; j < D; j += 32) {
    const float v = a[j] + gb[j] + r[j];
    s += v; s2 += v * v;
  }
#pragma unroll
  for (int o = 16; o > 0; o >>= 1) { s += __shfl_xor(s, o, 32); s2 += __shfl_xor(s2, o, 32); }
  const float mean = s / D;
  const float var = s2 / D - mean * mean;
  const float inv = rsqrtf(var + 1e-5f);
  for (int j = lane; j < D; j += 32) {
    const float v = a[j] + gb[j] + r[j];
    const float y = (v - mean) * inv * g[j] + b[j];
    out[(size_t)row * D + j] = siluf(y);
  }
}

__global__ void seg_max_kernel(const float* __restrict__ v, const int* __restrict__ ids,
                               float* __restrict__ smax, int n) {
  int t = blockIdx.x * blockDim.x + threadIdx.x;
  if (t < n) atomicMaxF(&smax[ids[t]], v[t]);
}

__global__ void seg_exp_kernel(const float* __restrict__ v, const float* __restrict__ smax,
                               const int* __restrict__ ids,
                               float* __restrict__ ex, float* __restrict__ ssum, int n) {
  int t = blockIdx.x * blockDim.x + threadIdx.x;
  if (t >= n) return;
  const float e = __expf(v[t] - smax[ids[t]]);
  ex[t] = e;
  atomicAdd(&ssum[ids[t]], e);
}

__global__ void pool_scatter_kernel(const float* __restrict__ h, const float* __restrict__ ex,
                                    const float* __restrict__ ssum, const int* __restrict__ ids,
                                    float* __restrict__ gemb, int N, int D) {
  const size_t total = (size_t)N * D;
  for (size_t idx = (size_t)blockIdx.x * blockDim.x + threadIdx.x; idx < total;
       idx += (size_t)gridDim.x * blockDim.x) {
    const int n = (int)(idx / D);
    const int c = (int)(idx % D);
    const float w = ex[n] / ssum[ids[n]];
    atomicAdd(&gemb[(size_t)ids[n] * D + c], w * h[idx]);
  }
}

__global__ void vae_kernel(const float* __restrict__ mu, const float* __restrict__ lv,
                           const float* __restrict__ eps,
                           float* __restrict__ logvar_out, float* __restrict__ z_out,
                           float* __restrict__ z_ws, int n) {
  int i = blockIdx.x * blockDim.x + threadIdx.x;
  if (i >= n) return;
  float l = fminf(10.f, fmaxf(-10.f, lv[i]));
  float s = fminf(1e6f, fmaxf(1e-6f, __expf(0.5f * l)));
  const float z = mu[i] + eps[i] * s;
  logvar_out[i] = l; z_out[i] = z; z_ws[i] = z;
}

__global__ void add_gather_kernel(const float* __restrict__ zproj, const int* __restrict__ gid,
                                  const float* __restrict__ ne, float* __restrict__ out,
                                  int N, int D) {
  const size_t total = (size_t)N * D;
  for (size_t idx = (size_t)blockIdx.x * blockDim.x + threadIdx.x; idx < total;
       idx += (size_t)gridDim.x * blockDim.x) {
    const int n = (int)(idx / D);
    const int c = (int)(idx % D);
    out[idx] = zproj[(size_t)gid[n] * D + c] + ne[idx];
  }
}

__global__ void gidx_kernel(const int* __restrict__ gids, const int* __restrict__ src,
                            const int* __restrict__ dst, int* __restrict__ gsrc,
                            int* __restrict__ gdst, int E) {
  int t = blockIdx.x * blockDim.x + threadIdx.x;
  if (t >= E) return;
  gsrc[t] = gids[src[t]];
  gdst[t] = gids[dst[t]];
}

// ---------------------------------------------------------------------------
// Host orchestration
// ---------------------------------------------------------------------------
static inline unsigned gblk(size_t total) {
  size_t b = (total + 255) / 256;
  if (b > (1u << 20)) b = (1u << 20);
  return (unsigned)(b ? b : 1);
}

struct GatP {
  const float *attn_l, *attn_r;
  const float *em1W, *em1b, *em2W, *em2b;
  const float *fcW, *gat_bias, *ln_b, *ln_g;
  const float *projW, *projb, *resW, *resb;
};

extern "C" void kernel_launch(void* const* d_in, const int* in_sizes, int n_in,
                              void* d_out, int out_size, void* d_ws, size_t ws_size,
                              hipStream_t stream) {
  (void)in_sizes; (void)n_in; (void)out_size; (void)ws_size;

  // --- inputs (setup_inputs insertion order) ---
  const float* node_feats = (const float*)d_in[0];   // [NN, 4]
  const float* edge_feats = (const float*)d_in[1];   // [NE, 3]
  const int*   src        = (const int*)d_in[2];
  const int*   dst        = (const int*)d_in[3];
  const int*   graph_ids  = (const int*)d_in[4];
  const float* eps        = (const float*)d_in[5];   // [NG, 64]

  // --- params: JAX pytree leaf order (alphabetical dict keys at every level) ---
  int pi = 6;
  auto P = [&](void) { return (const float*)d_in[pi++]; };
  auto loadGat = [&](void) {
    GatP g;
    g.attn_l = P(); g.attn_r = P();
    g.em1W = P(); g.em1b = P(); g.em2W = P(); g.em2b = P();
    g.fcW = P(); g.gat_bias = P(); g.ln_b = P(); g.ln_g = P();
    g.projW = P(); g.projb = P(); g.resW = P(); g.resb = P();
    return g;
  };
  const float *edge_dec1_W = P(), *edge_dec1_b = P();
  const float *edge_dec2_W = P(), *edge_dec2_b = P();
  const float *energy1_W = P(), *energy1_b = P();
  const float *energy2_W = P(), *energy2_b = P();
  const float *fc_logvar_W = P(), *fc_logvar_b = P();
  const float *fc_mu_W = P(), *fc_mu_b = P();
  GatP gat1 = loadGat();
  GatP gat2 = loadGat();
  GatP gat3 = loadGat();
  const float *gate1_W = P(), *gate1_b = P();
  const float *gate2_W = P(), *gate2_b = P();
  const float *latent_proj_W = P(), *latent_proj_b = P();
  const float *node_dec1_W = P(), *node_dec1_b = P();
  const float *node_dec2_W = P(), *node_dec2_b = P();
  const float *node_emb_proj_W = P(), *node_emb_proj_b = P();
  const float *stress1_W = P(), *stress1_b = P();
  const float *stress2_W = P(), *stress2_b = P();

  // --- workspace bump allocator ---
  float* wsf = (float*)d_ws;
  size_t off = 0;
  auto alloc = [&](size_t nElts) {
    float* p = wsf + off;
    off += (nElts + 255) & ~(size_t)255;
    return p;
  };
  float* hA    = alloc((size_t)NN * FDIM);
  float* hB    = alloc((size_t)NN * FDIM);
  float* proj  = alloc((size_t)NN * 512);
  float* temp  = alloc((size_t)NN * 520);
  float* hh    = alloc((size_t)NN * FDIM);
  float* resid = alloc((size_t)NN * FDIM);
  float* aggr  = alloc((size_t)NN * FDIM);
  float* ef2   = alloc((size_t)NE * EFEAT);
  float* el    = alloc((size_t)NN * HEADS);
  float* er    = alloc((size_t)NN * HEADS);
  float* eatt  = alloc((size_t)NE * HEADS);
  float* eexp  = alloc((size_t)NE * HEADS);
  float* segmax = alloc((size_t)NN * HEADS);
  float* segsum = alloc((size_t)NN * HEADS);
  float* gate_h = alloc((size_t)NN * HIDD);
  float* gval   = alloc((size_t)NN);
  float* gexp   = alloc((size_t)NN);
  float* gmax   = alloc((size_t)NG);
  float* gsum   = alloc((size_t)NG);
  float* gemb   = alloc((size_t)NG * FDIM);
  float* lvbuf  = alloc((size_t)NG * LATD);
  float* zbuf   = alloc((size_t)NG * LATD);
  float* zproj  = alloc((size_t)NG * 2 * HIDD);
  float* nebuf  = alloc((size_t)NN * 2 * HIDD);
  float* decin  = alloc((size_t)NN * 2 * HIDD);
  float* nh1    = alloc((size_t)NN * HIDD);
  float* eh1    = alloc((size_t)NE * HIDD);
  float* eng_h  = alloc((size_t)NG * HIDD);
  float* str_h  = alloc((size_t)NG * HIDD);
  int* gsrc = (int*)alloc((size_t)NE);
  int* gdst = (int*)alloc((size_t)NE);
  __bf16* bfA = (__bf16*)alloc((size_t)12 * 1024 * 1024);  // 24M bf16 (max: eh1 20.5M)
  __bf16* bfW = (__bf16*)alloc((size_t)512 * 1024);        // 1M bf16 (max: fc_W 527K)

  // GEMM wrapper: convert operands to bf16 once, then WMMA GEMM
  auto G = [&](const float* A, const float* W, const float* bias, float* C,
               int M, int N, int K, int act, size_t aElems,
               int lda = 0, const int* iA = nullptr, const int* iB = nullptr, int Ks = 0) {
    const size_t wElems = (size_t)K * N;
    cvt_bf16_kernel<<<gblk((aElems + 7) / 8), 256, 0, stream>>>(A, bfA, aElems);
    cvt_bf16_kernel<<<gblk((wElems + 7) / 8), 256, 0, stream>>>(W, bfW, wElems);
    dim3 grid((M + 63) / 64, (N + 63) / 64);
    gemm_wmma_kernel<<<grid, 256, 0, stream>>>(bfA, bfW, bias, C, M, N, K, iA, iB, Ks, lda, act);
  };

  // --- one GAT block ---
  auto gat_layer = [&](const GatP& g, const float* h_in, int din, int dout, float* h_out) {
    const int HD = HEADS * dout;
    const int Wt = din + EFEAT;
    const int TS = (Wt + 7) & ~7;   // padded temp row stride (keeps A loads aligned)
    // residual + node projection (WMMA GEMMs)
    G(h_in, g.resW, g.resb, resid, NN, HD, din, 0, (size_t)NN * din);
    G(h_in, g.projW, g.projb, proj, NN, din, din, 0, (size_t)NN * din);
    // edge MLP
    edge_mlp_kernel<<<gblk(NE), 256, 0, stream>>>(edge_feats, g.em1W, g.em1b, g.em2W, g.em2b, ef2, NE);
    // temp = seg_sum(src_in, src) + seg_sum(dst_in, dst)
    fill_kernel<<<gblk((size_t)NN * TS), 256, 0, stream>>>(temp, 0.f, (size_t)NN * TS);
    scatter_temp_kernel<<<gblk((size_t)NE * Wt), 256, 0, stream>>>(proj, ef2, src, dst, temp, NE, din, TS);
    // hh = temp @ fc_W   (no bias; A row stride = TS)
    G(temp, g.fcW, nullptr, hh, NN, HD, Wt, 0, (size_t)NN * TS, TS);
    // attention logits
    attn_lr_kernel<<<gblk((size_t)NN * HEADS), 256, 0, stream>>>(hh, g.attn_l, g.attn_r, el, er, NN, dout);
    fill_kernel<<<gblk((size_t)NN * HEADS), 256, 0, stream>>>(segmax, -1e30f, (size_t)NN * HEADS);
    fill_kernel<<<gblk((size_t)NN * HEADS), 256, 0, stream>>>(segsum, 0.f, (size_t)NN * HEADS);
    edge_score_kernel<<<gblk((size_t)NE * HEADS), 256, 0, stream>>>(el, er, src, dst, eatt, segmax, NE);
    edge_exp_kernel<<<gblk((size_t)NE * HEADS), 256, 0, stream>>>(eatt, segmax, dst, eexp, segsum, NE);
    // weighted aggregation
    fill_kernel<<<gblk((size_t)NN * HD), 256, 0, stream>>>(aggr, 0.f, (size_t)NN * HD);
    attn_scatter_kernel<<<gblk((size_t)NE * HD), 256, 0, stream>>>(hh, eexp, segsum, src, dst, aggr, NE, dout);
    // out = silu(LN(aggr + gat_bias + resid))
    ln_silu_kernel<<<(NN + 3) / 4, 128, 0, stream>>>(aggr, resid, g.gat_bias, g.ln_g, g.ln_b, h_out, NN, HD);
  };

  // --- encoder ---
  gat_layer(gat1, node_feats, NFEAT, HIDD / 2, hA);          // -> [NN, 256]
  gat_layer(gat2, hA, HEADS * (HIDD / 2), HIDD, hB);         // -> [NN, 512]
  gat_layer(gat3, hB, HEADS * HIDD, 2 * HIDD, hA);           // -> [NN, 1024]
  const float* h = hA;

  // --- global attention pooling ---
  G(h, gate1_W, gate1_b, gate_h, NN, HIDD, FDIM, 1, (size_t)NN * FDIM);
  G(gate_h, gate2_W, gate2_b, gval, NN, 1, HIDD, 0, (size_t)NN * HIDD);
  fill_kernel<<<1, 256, 0, stream>>>(gmax, -1e30f, NG);
  fill_kernel<<<1, 256, 0, stream>>>(gsum, 0.f, NG);
  seg_max_kernel<<<gblk(NN), 256, 0, stream>>>(gval, graph_ids, gmax, NN);
  seg_exp_kernel<<<gblk(NN), 256, 0, stream>>>(gval, gmax, graph_ids, gexp, gsum, NN);
  fill_kernel<<<gblk((size_t)NG * FDIM), 256, 0, stream>>>(gemb, 0.f, (size_t)NG * FDIM);
  pool_scatter_kernel<<<gblk((size_t)NN * FDIM), 256, 0, stream>>>(h, gexp, gsum, graph_ids, gemb, NN, FDIM);

  // --- VAE latent ---
  float* out = (float*)d_out;
  const size_t O_MU = 0, O_LV = 6400, O_Z = 12800, O_RN = 19200, O_RE = 59200,
               O_EN = 539200, O_ST = 539400;
  G(gemb, fc_mu_W, fc_mu_b, out + O_MU, NG, LATD, FDIM, 0, (size_t)NG * FDIM);
  G(gemb, fc_logvar_W, fc_logvar_b, lvbuf, NG, LATD, FDIM, 0, (size_t)NG * FDIM);
  vae_kernel<<<gblk((size_t)NG * LATD), 256, 0, stream>>>(out + O_MU, lvbuf, eps,
                                                          out + O_LV, out + O_Z, zbuf, NG * LATD);

  // --- decoder: nodes ---
  G(zbuf, latent_proj_W, latent_proj_b, zproj, NG, 2 * HIDD, LATD, 1, (size_t)NG * LATD);
  G(h, node_emb_proj_W, node_emb_proj_b, nebuf, NN, 2 * HIDD, FDIM, 1, (size_t)NN * FDIM);
  add_gather_kernel<<<gblk((size_t)NN * 2 * HIDD), 256, 0, stream>>>(zproj, graph_ids, nebuf, decin, NN, 2 * HIDD);
  G(decin, node_dec1_W, node_dec1_b, nh1, NN, HIDD, 2 * HIDD, 1, (size_t)NN * 2 * HIDD);
  G(nh1, node_dec2_W, node_dec2_b, out + O_RN, NN, NFEAT, HIDD, 0, (size_t)NN * HIDD);

  // --- decoder: edges (fused gather-concat GEMM: A row = [zproj[gid[src]], zproj[gid[dst]]]) ---
  gidx_kernel<<<gblk(NE), 256, 0, stream>>>(graph_ids, src, dst, gsrc, gdst, NE);
  G(zproj, edge_dec1_W, edge_dec1_b, eh1, NE, HIDD, 4 * HIDD, 1, (size_t)NG * 2 * HIDD,
    2 * HIDD, gsrc, gdst, 2 * HIDD);
  G(eh1, edge_dec2_W, edge_dec2_b, out + O_RE, NE, EFEAT, HIDD, 0, (size_t)NE * HIDD);

  // --- energy / stress heads ---
  G(zbuf, energy1_W, energy1_b, eng_h, NG, HIDD, LATD, 1, (size_t)NG * LATD);
  G(eng_h, energy2_W, energy2_b, out + O_EN, NG, 2, HIDD, 0, (size_t)NG * HIDD);
  G(zbuf, stress1_W, stress1_b, str_h, NG, HIDD, LATD, 1, (size_t)NG * LATD);
  G(str_h, stress2_W, stress2_b, out + O_ST, NG, 9, HIDD, 0, (size_t)NG * HIDD);
}
